// LSTMModel_77532749627606
// MI455X (gfx1250) — matrix operand order repaired and checked by reading, not compiled, not run on hardware
//
#include <hip/hip_runtime.h>
#include <math.h>

#define Bsz   256
#define Tlen  512
#define Demb  512
#define Hdim  1024
#define G4H   4096
#define Vocab 5000
#define VocabPad 5120              // FC weight rows padded to a 128-col tile multiple

typedef __attribute__((ext_vector_type(4)))  __bf16 v4bf;
typedef __attribute__((ext_vector_type(8)))  __bf16 v8bf;
typedef __attribute__((ext_vector_type(16))) __bf16 v16bf;
typedef __attribute__((ext_vector_type(4)))  float  v4f;
typedef __attribute__((ext_vector_type(8)))  float  v8f;

#define LDS_STRIDE 40              // 32 K padded to 40 bf16 (bank-conflict-free b128 reads)
#define LDS_TILE   (64 * LDS_STRIDE)

// Global-address-space 16-byte bf16 chunk (forces global_load_b128, keeps DScnt clean)
typedef const __attribute__((address_space(1))) v8bf* gv8bf_p;

// Build a 16-element bf16 fragment from two 16-byte GLOBAL chunks.
static __device__ __forceinline__ v16bf ld_frag_g(const __bf16* p0, const __bf16* p1) {
  v8bf a = *(gv8bf_p)p0;
  v8bf b = *(gv8bf_p)p1;
  return __builtin_shufflevector(a, b, 0,1,2,3,4,5,6,7,8,9,10,11,12,13,14,15);
}
// Same from generic (LDS) pointers.
static __device__ __forceinline__ v16bf ld_frag(const __bf16* p0, const __bf16* p1) {
  v8bf a = *(const v8bf*)p0;
  v8bf b = *(const v8bf*)p1;
  return __builtin_shufflevector(a, b, 0,1,2,3,4,5,6,7,8,9,10,11,12,13,14,15);
}

#define WMMA_BF16(A, B, C) \
  __builtin_amdgcn_wmma_f32_16x16x32_bf16(false, (A), false, (B), (short)0, (C), false, false)

static __device__ __forceinline__ float fast_sigmoid(float x) {
  return __builtin_amdgcn_rcpf(1.0f + __expf(-x));          // v_exp + v_rcp (TRANS)
}
static __device__ __forceinline__ float fast_tanh(float x) {
  return 2.0f * __builtin_amdgcn_rcpf(1.0f + __expf(-2.0f * x)) - 1.0f;
}

// ---------------------------------------------------------------------------
// Shared WMMA main loop, templated on segment depths K0/K1 so each pair's
// 8 B-fragment loads use ONE base pointer + compile-time immediate offsets
// (s*16*K*2 bytes <= 98304, inside signed-24-bit IOFFSET). Segment boundary is
// peeled. Interior pair-loops are kept ROLLED (#pragma unroll 1) so the
// accumulators stay in loop-carried registers: no RA copies, no hazard NOPs.
//   * LDS double-buffered 64x32 A-tiles (one barrier per chunk)
//   * B loads for chunk c+1 issue before chunk c's WMMAs (latency hiding)
//   * wave tile 32(M) x 64(N): 2 A-frags x 4 B-frags -> 8 WMMAs per chunk
// Block: 128 threads = 4 waves (2 wave-rows x 2 wave-cols) -> 64x128 block tile.
// ---------------------------------------------------------------------------
template <int K0, int K1>
static __device__ __forceinline__ void wmma_mainloop(
    const __bf16* __restrict__ A0, int lda0, const __bf16* __restrict__ W0,
    const __bf16* __restrict__ A1, int lda1, const __bf16* __restrict__ W1,
    __bf16* lds, int rowBase, int col0, int tid, v8f (&acc)[2][4]) {
  const int lane    = tid & 31;
  const int waveRow = (tid >> 6) & 1;      // wave 0,1 -> row 0; wave 2,3 -> row 1
  const int half    = lane >> 4;           // 0: lanes 0-15, 1: lanes 16-31
  const int lrow    = lane & 15;
  const int sr      = tid >> 1;            // staging row 0..63
  const int sc      = (tid & 1) * 16;      // staging k-offset 0/16

  constexpr int n0 = K0 / 32;              // even
  constexpr int n1 = K1 / 32;              // even or 0

  auto stageAt = [&](const __bf16* p, int buf) {
    v8bf lo = *(gv8bf_p)p;
    v8bf hi = *(gv8bf_p)(p + 8);
    __bf16* dst = &lds[buf * LDS_TILE + sr * LDS_STRIDE + sc];
    *(v8bf*)dst       = lo;
    *(v8bf*)(dst + 8) = hi;
    __builtin_prefetch(p + 32, 0, 1);      // global_prefetch_b8: next chunk
  };
  // 4 B fragments from ONE base pointer; strideK/off are compile-time at all sites
  auto loadB4 = [&](v16bf (&dst)[4], const __bf16* w, int strideK, int off) {
#pragma unroll
    for (int s = 0; s < 4; ++s)
      dst[s] = ld_frag_g(w + (size_t)s * 16 * strideK + off,
                         w + (size_t)s * 16 * strideK + off + 8);
  };
  auto mma_chunk = [&](int buf, v16bf (&bf)[4]) {
    const __bf16* ap0 = &lds[buf * LDS_TILE + (waveRow * 32 + lrow) * LDS_STRIDE + half * 16];
    const __bf16* ap1 = ap0 + 16 * LDS_STRIDE;
    v16bf a0 = ld_frag(ap0, ap0 + 8);
    v16bf a1 = ld_frag(ap1, ap1 + 8);
#pragma unroll
    for (int s = 0; s < 4; ++s) {
      acc[0][s] = WMMA_BF16(a0, bf[s], acc[0][s]);
      acc[1][s] = WMMA_BF16(a1, bf[s], acc[1][s]);
    }
  };

  // per-segment base pointers (lane-dependent parts computed ONCE)
  const __bf16* a0p = A0 + (size_t)(rowBase + sr) * lda0 + sc;
  const __bf16* a1p = A1 + (size_t)(rowBase + sr) * lda1 + sc;
  const __bf16* w0  = W0 + (size_t)col0 * K0 + half * 16;
  const __bf16* w1  = W1 + (size_t)col0 * K1 + half * 16;

  v16bf bA[4], bB[4];

  // prologue: stage chunk 0 into buf0, preload its B fragments
  stageAt(a0p, 0);
  loadB4(bA, w0, K0, 0);
  __syncthreads();

  // ---- segment 0 interior pairs (next-even chunk still in segment 0) ----
#pragma unroll 1
  for (int c = 0; c + 2 < n0; c += 2) {
    stageAt(a0p + 32, 1);
    loadB4(bB, w0, K0, 32);
    mma_chunk(0, bA);
    __syncthreads();
    stageAt(a0p + 64, 0);
    loadB4(bA, w0, K0, 64);
    a0p += 64;
    w0  += 64;
    mma_chunk(1, bB);
    __syncthreads();
  }

  // ---- segment 0 boundary pair (chunks n0-2, n0-1): next-even is seg 1 ----
  stageAt(a0p + 32, 1);
  loadB4(bB, w0, K0, 32);
  mma_chunk(0, bA);
  __syncthreads();
  if (n1 > 0) {
    stageAt(a1p, 0);
    loadB4(bA, w1, K1, 0);
  }
  mma_chunk(1, bB);
  __syncthreads();

  // ---- segment 1 ----
  if (n1 > 0) {
#pragma unroll 1
    for (int c = 0; c + 2 < n1; c += 2) {
      stageAt(a1p + 32, 1);
      loadB4(bB, w1, K1, 32);
      mma_chunk(0, bA);
      __syncthreads();
      stageAt(a1p + 64, 0);
      loadB4(bA, w1, K1, 64);
      a1p += 64;
      w1  += 64;
      mma_chunk(1, bB);
      __syncthreads();
    }
    // final pair (compute-only tail after staging its odd chunk)
    stageAt(a1p + 32, 1);
    loadB4(bB, w1, K1, 32);
    mma_chunk(0, bA);
    __syncthreads();
    mma_chunk(1, bB);
    __syncthreads();
  }
}

// ---------------- embedding + positional add -> bf16 x0 [T][B][Demb] ----------
__global__ void embed_kernel(const int* __restrict__ inputs,
                             const float* __restrict__ etok,
                             const float* __restrict__ epos,
                             __bf16* __restrict__ x0) {
  size_t q = (size_t)blockIdx.x * blockDim.x + threadIdx.x;   // element-quad index
  if (q >= (size_t)Tlen * Bsz * (Demb / 4)) return;
  int d4 = (int)(q % (Demb / 4));
  size_t tb = q / (Demb / 4);
  int b = (int)(tb % Bsz);
  int t = (int)(tb / Bsz);
  int tok = inputs[(size_t)b * Tlen + t];
  const v4f e = *(const v4f*)&etok[(size_t)tok * Demb + d4 * 4];
  const v4f p = *(const v4f*)&epos[(size_t)t * Demb + d4 * 4];
  v4bf o;
#pragma unroll
  for (int u = 0; u < 4; ++u) o[u] = (__bf16)(e[u] + p[u]);
  *(v4bf*)&x0[q * 4] = o;
}

// ---------------- fp32 -> bf16 conversion (x4 vectorized) ---------------------
__global__ void f32_to_bf16(const float* __restrict__ src, __bf16* __restrict__ dst,
                            size_t n4) {
  size_t i = (size_t)blockIdx.x * blockDim.x + threadIdx.x;
  if (i >= n4) return;
  const v4f v = *(const v4f*)&src[i * 4];
  v4bf o;
#pragma unroll
  for (int u = 0; u < 4; ++u) o[u] = (__bf16)v[u];
  *(v4bf*)&dst[i * 4] = o;
}

// ---------------- bias sum ----------------------------------------------------
__global__ void bias_sum(const float* __restrict__ bi, const float* __restrict__ bh,
                         float* __restrict__ bs) {
  int i = blockIdx.x * blockDim.x + threadIdx.x;
  if (i < G4H) bs[i] = bi[i] + bh[i];
}

// ---------------- zero h (bf16) and c (fp32) ---------------------------------
__global__ void zero_state(__bf16* __restrict__ h, float* __restrict__ c) {
  int i = blockIdx.x * blockDim.x + threadIdx.x;
  if (i < Bsz * Hdim) { h[i] = (__bf16)0.0f; c[i] = 0.0f; }
}

// ---------------- fused per-timestep GEMM: gates = x_t@Wi^T + h@Wh^T ----------
template <int KX>
__global__ __launch_bounds__(128)
void lstm_step_gemm(const __bf16* __restrict__ xt,
                    const __bf16* __restrict__ Wi,
                    const __bf16* __restrict__ hprev,
                    const __bf16* __restrict__ Wh,
                    float* __restrict__ gates) {
  __shared__ __bf16 lds[2 * LDS_TILE];

  const int tid     = threadIdx.x;
  const int lane    = tid & 31;
  const int wave    = tid >> 5;
  const int waveRow = wave >> 1;
  const int waveCol = wave & 1;
  const int rowBase = blockIdx.y * 64;
  const int colBase = blockIdx.x * 128 + waveCol * 64;
  const int half    = lane >> 4;
  const int lrow    = lane & 15;
  const int col0    = colBase + lrow;

  v8f acc[2][4] = {};
  wmma_mainloop<KX, Hdim>(xt, KX, Wi, hprev, Hdim, Wh,
                          lds, rowBase, col0, tid, acc);

  // C/D layout: lanes 0-15: N=lane, M=e; lanes 16-31: N=lane-16, M=e+8
#pragma unroll
  for (int sub = 0; sub < 2; ++sub) {
    const int orow = rowBase + waveRow * 32 + sub * 16 + half * 8;
    float* g0 = &gates[(size_t)orow * G4H + col0];
#pragma unroll
    for (int e = 0; e < 8; ++e) {
      g0[(size_t)e * G4H +  0] = acc[sub][0][e];
      g0[(size_t)e * G4H + 16] = acc[sub][1][e];
      g0[(size_t)e * G4H + 32] = acc[sub][2][e];
      g0[(size_t)e * G4H + 48] = acc[sub][3][e];
    }
  }
}

// ---------------- LSTM pointwise: gates -> (c,h), emit bf16 h (x4) ------------
__global__ void lstm_pointwise(const float* __restrict__ gates,
                               const float* __restrict__ bsum,
                               float* __restrict__ c,
                               __bf16* __restrict__ hbf,
                               __bf16* __restrict__ xout) {
  int q = blockIdx.x * blockDim.x + threadIdx.x;   // < B*H/4
  int b = q >> 8;                                  // /(H/4)
  int j = (q & 255) * 4;
  const float* gr = gates + (size_t)b * G4H;
  const v4f gi = *(const v4f*)&gr[j];
  const v4f gf = *(const v4f*)&gr[Hdim + j];
  const v4f gg = *(const v4f*)&gr[2 * Hdim + j];
  const v4f go = *(const v4f*)&gr[3 * Hdim + j];
  const v4f bi = *(const v4f*)&bsum[j];
  const v4f bf = *(const v4f*)&bsum[Hdim + j];
  const v4f bg = *(const v4f*)&bsum[2 * Hdim + j];
  const v4f bo = *(const v4f*)&bsum[3 * Hdim + j];
  v4f cc = *(const v4f*)&c[b * Hdim + j];

  v4f cn;
  v4bf hb;
#pragma unroll
  for (int u = 0; u < 4; ++u) {
    float iv = fast_sigmoid(gi[u] + bi[u]);
    float fv = fast_sigmoid(gf[u] + bf[u]);
    float gv = fast_tanh(gg[u] + bg[u]);
    float ov = fast_sigmoid(go[u] + bo[u]);
    cn[u] = fv * cc[u] + iv * gv;
    hb[u] = (__bf16)(ov * fast_tanh(cn[u]));
  }
  *(v4f*)&c[b * Hdim + j] = cn;
  *(v4bf*)&hbf[b * Hdim + j]  = hb;
  *(v4bf*)&xout[b * Hdim + j] = hb;   // xout pre-offset to timestep slice
}

// ---------------- final FC: out[B,V] = h @ fcW^T + fc_b -----------------------
// fcW is padded to VocabPad rows, so no column clamping on loads; stores masked.
__global__ __launch_bounds__(128)
void fc_gemm(const __bf16* __restrict__ h, const __bf16* __restrict__ W,
             const float* __restrict__ bias, float* __restrict__ out) {
  __shared__ __bf16 lds[2 * LDS_TILE];

  const int tid     = threadIdx.x;
  const int lane    = tid & 31;
  const int wave    = tid >> 5;
  const int waveRow = wave >> 1;
  const int waveCol = wave & 1;
  const int rowBase = blockIdx.y * 64;
  const int colBase = blockIdx.x * 128 + waveCol * 64;
  const int half    = lane >> 4;
  const int lrow    = lane & 15;
  const int col0    = colBase + lrow;

  v8f acc[2][4] = {};
  wmma_mainloop<Hdim, 0>(h, Hdim, W, h, Hdim, W,   // single segment
                         lds, rowBase, col0, tid, acc);

#pragma unroll
  for (int sub = 0; sub < 2; ++sub) {
    const int orow = rowBase + waveRow * 32 + sub * 16 + half * 8;
#pragma unroll
    for (int s = 0; s < 4; ++s) {
      int cl = col0 + s * 16;
      if (cl >= Vocab) continue;
      float bb = bias[cl];
#pragma unroll
      for (int e = 0; e < 8; ++e)
        out[(size_t)(orow + e) * Vocab + cl] = acc[sub][s][e] + bb;
    }
  }
}

// ---------------------------------------------------------------------------
static inline size_t align256(size_t x) { return (x + 255) & ~(size_t)255; }

extern "C" void kernel_launch(void* const* d_in, const int* in_sizes, int n_in,
                              void* d_out, int out_size, void* d_ws, size_t ws_size,
                              hipStream_t stream) {
  (void)in_sizes; (void)n_in; (void)out_size; (void)ws_size;

  const int*   inputs  = (const int*)  d_in[0];
  const float* emb_tok = (const float*)d_in[1];
  const float* emb_pos = (const float*)d_in[2];
  const float* fc_W    = (const float*)d_in[3];
  const float* fc_b    = (const float*)d_in[4];
  const float* W_ih[3] = { (const float*)d_in[5],  (const float*)d_in[9],  (const float*)d_in[13] };
  const float* W_hh[3] = { (const float*)d_in[6],  (const float*)d_in[10], (const float*)d_in[14] };
  const float* b_ih[3] = { (const float*)d_in[7],  (const float*)d_in[11], (const float*)d_in[15] };
  const float* b_hh[3] = { (const float*)d_in[8],  (const float*)d_in[12], (const float*)d_in[16] };
  float* out = (float*)d_out;

  // workspace carve
  char* ws = (char*)d_ws;
  size_t off = 0;
  auto carve = [&](size_t bytes) -> void* { void* p = ws + off; off = align256(off + bytes); return p; };

  __bf16* xA     = (__bf16*)carve((size_t)Tlen * Bsz * Hdim * sizeof(__bf16));
  __bf16* xB     = (__bf16*)carve((size_t)Tlen * Bsz * Hdim * sizeof(__bf16));
  __bf16* Wi_bf  = (__bf16*)carve((size_t)G4H * Hdim * sizeof(__bf16));
  __bf16* Wh_bf  = (__bf16*)carve((size_t)G4H * Hdim * sizeof(__bf16));
  __bf16* fcW_bf = (__bf16*)carve((size_t)VocabPad * Hdim * sizeof(__bf16));  // padded rows
  float*  gates  = (float*) carve((size_t)Bsz * G4H * sizeof(float));
  float*  bsum   = (float*) carve((size_t)G4H * sizeof(float));
  __bf16* h_bf   = (__bf16*)carve((size_t)Bsz * Hdim * sizeof(__bf16));
  float*  cstate = (float*) carve((size_t)Bsz * Hdim * sizeof(float));

  // 1. embedding + positional add -> xA [T][B][512] (bf16)
  {
    size_t n4 = (size_t)Tlen * Bsz * (Demb / 4);
    embed_kernel<<<dim3((unsigned)((n4 + 255) / 256)), 256, 0, stream>>>(inputs, emb_tok, emb_pos, xA);
  }
  // 2. fc weights -> bf16 (once; only the real Vocab rows)
  {
    size_t n4 = (size_t)Vocab * Hdim / 4;
    f32_to_bf16<<<dim3((unsigned)((n4 + 255) / 256)), 256, 0, stream>>>(fc_W, fcW_bf, n4);
  }

  // 3. layers
  for (int l = 0; l < 3; ++l) {
    const int Din = (l == 0) ? Demb : Hdim;
    const __bf16* xin  = (l == 1) ? xB : xA;   // l0: xA, l1: xB, l2: xA
    __bf16*       xout = (l == 1) ? xA : xB;   // l0 -> xB, l1 -> xA, l2 -> xB

    size_t nwi4 = (size_t)G4H * Din / 4;
    f32_to_bf16<<<dim3((unsigned)((nwi4 + 255) / 256)), 256, 0, stream>>>(W_ih[l], Wi_bf, nwi4);
    size_t nwh4 = (size_t)G4H * Hdim / 4;
    f32_to_bf16<<<dim3((unsigned)((nwh4 + 255) / 256)), 256, 0, stream>>>(W_hh[l], Wh_bf, nwh4);
    bias_sum<<<dim3(G4H / 256), 256, 0, stream>>>(b_ih[l], b_hh[l], bsum);
    zero_state<<<dim3(Bsz * Hdim / 256), 256, 0, stream>>>(h_bf, cstate);

    const dim3 grid(G4H / 128, Bsz / 64);
    for (int t = 0; t < Tlen; ++t) {
      if (l == 0) {
        lstm_step_gemm<Demb><<<grid, 128, 0, stream>>>(
            xin + (size_t)t * Bsz * Demb, Wi_bf, h_bf, Wh_bf, gates);
      } else {
        lstm_step_gemm<Hdim><<<grid, 128, 0, stream>>>(
            xin + (size_t)t * Bsz * Hdim, Wi_bf, h_bf, Wh_bf, gates);
      }
      lstm_pointwise<<<dim3(Bsz * Hdim / 4 / 256), 256, 0, stream>>>(
          gates, bsum, cstate, h_bf, xout + (size_t)t * Bsz * Hdim);
    }
  }

  // 4. final FC from h at t = T-1 of layer 2
  fc_gemm<<<dim3(VocabPad / 128, Bsz / 64), 128, 0, stream>>>(h_bf, fcW_bf, fc_b, out);
}
